// VinaScoreHead_53824530154140
// MI455X (gfx1250) — matrix operand order repaired and checked by reading, not compiled, not run on hardware
//
#include <hip/hip_runtime.h>
#include <hip/hip_bf16.h>

typedef __attribute__((ext_vector_type(4)))  float    f4;
typedef __attribute__((ext_vector_type(8)))  float    v8f;
typedef __attribute__((ext_vector_type(8)))  _Float16 h8;
typedef __attribute__((ext_vector_type(16))) _Float16 v16h;

union V16 { v16h v; f4 q[2]; };
union H8  { h8 h;  f4 q; };

#define NB 4
#define NN 256
#define DM 128
#define NTYPES 29
#define HALF_LOG_2PI 0.9189385332046727f
#define NEG_INF -1e9f
#define TOTAL_PAIRS (NB * NN * NN)

// ---------------- helpers ----------------
__device__ __forceinline__ float redsum16(float v) {
    v += __shfl_xor(v, 8, 16);
    v += __shfl_xor(v, 4, 16);
    v += __shfl_xor(v, 2, 16);
    v += __shfl_xor(v, 1, 16);
    return v;
}
__device__ __forceinline__ float redsum32(float v) {
    v += __shfl_xor(v, 16, 32);
    v += __shfl_xor(v, 8, 32);
    v += __shfl_xor(v, 4, 32);
    v += __shfl_xor(v, 2, 32);
    v += __shfl_xor(v, 1, 32);
    return v;
}
__device__ __forceinline__ float eluf(float x) { return x > 0.f ? x : (__expf(x) - 1.f); }

// ---------------- node LN precompute: nf = LN(node_feats[:,1:,:]) ----------------
// block = 256 threads = 8 waves; each wave handles one node row (128 feats, 4/lane)
__global__ __launch_bounds__(256) void nf_kernel(const float* __restrict__ node_feats,
                                                 const float* __restrict__ g,
                                                 const float* __restrict__ bb,
                                                 float* __restrict__ nf) {
    int w = threadIdx.x >> 5, L = threadIdx.x & 31;
    int row = blockIdx.x * 8 + w;              // 0 .. NB*NN-1
    int b = row >> 8, nd = row & 255;
    const float* x = node_feats + ((size_t)(b * 257 + 1 + nd)) * DM + L * 4;
    f4 xv = *(const f4*)x;
    float s  = xv[0] + xv[1] + xv[2] + xv[3];
    float ss = xv[0]*xv[0] + xv[1]*xv[1] + xv[2]*xv[2] + xv[3]*xv[3];
    s = redsum32(s); ss = redsum32(ss);
    float mean = s * (1.0f / DM);
    float var  = ss * (1.0f / DM) - mean * mean;
    float rs   = rsqrtf(var + 1e-6f);
    f4 gv = *(const f4*)(g  + L * 4);
    f4 bv = *(const f4*)(bb + L * 4);
    f4 o;
    for (int k = 0; k < 4; ++k) o[k] = (xv[k] - mean) * rs * gv[k] + bv[k];
    *(f4*)(nf + (size_t)row * DM + L * 4) = o;
}

// ---------------- weight pack: WMMA-B fragment layout, f32 -> f16 ----------------
// W1pack: per (head, ntile(16), kstep(4)): 32 lanes x 16 halves contiguous.
//   lane<16: N=lane,     K = kstep*32 + 0..15
//   lane>=16:N=lane-16,  K = kstep*32 + 16..31
// W2pack: per (head, kstep(8)): same, N padded 4->16 with zeros.
__global__ __launch_bounds__(256) void pack_kernel(const float* __restrict__ W1m, const float* __restrict__ W1s, const float* __restrict__ W1p,
                                                   const float* __restrict__ W2m, const float* __restrict__ W2s, const float* __restrict__ W2p,
                                                   _Float16* __restrict__ W1pack, _Float16* __restrict__ W2pack) {
    int tid = blockIdx.x * blockDim.x + threadIdx.x;
    if (tid < 3 * 16 * 4 * 32) {
        int lane = tid & 31;
        int kb   = (tid >> 5) & 3;
        int hd   = tid >> 11;
        const float* W = hd == 0 ? W1m : (hd == 1 ? W1s : W1p);
        int n     = ((tid >> 7) & 15) * 16 + (lane & 15);
        int kbase = kb * 32 + ((lane >= 16) ? 16 : 0);
        _Float16* dst = W1pack + (size_t)tid * 16;
        for (int h = 0; h < 16; ++h) dst[h] = (_Float16)W[(kbase + h) * 256 + n];
    } else if (tid < 3 * 16 * 4 * 32 + 3 * 8 * 32) {
        int u = tid - 3 * 16 * 4 * 32;
        int lane = u & 31;
        int kb   = (u >> 5) & 7;
        int hd   = u >> 8;
        const float* W = hd == 0 ? W2m : (hd == 1 ? W2s : W2p);
        int n     = lane & 15;
        int kbase = kb * 32 + ((lane >= 16) ? 16 : 0);
        _Float16* dst = W2pack + (size_t)u * 16;
        for (int h = 0; h < 16; ++h)
            dst[h] = (n < 4) ? (_Float16)W[(kbase + h) * 4 + n] : (_Float16)0.f;
    }
}

// ---------------- fused main kernel: 16-pair tile per workgroup ----------------
__global__ __launch_bounds__(256) void main_kernel(
    const float* __restrict__ pair_feats, const float* __restrict__ Dm,
    const float* __restrict__ yv, const int* __restrict__ node_type,
    const unsigned char* __restrict__ pair_mask,
    const float* __restrict__ pg, const float* __restrict__ pb,
    const float* __restrict__ fg, const float* __restrict__ fb,
    const float* __restrict__ b1m, const float* __restrict__ b1s, const float* __restrict__ b1p,
    const float* __restrict__ b2m, const float* __restrict__ b2s, const float* __restrict__ b2p,
    const float* __restrict__ vdw_t, const float* __restrict__ hydro_t, const float* __restrict__ hbond_t,
    const float* __restrict__ nf,
    const _Float16* __restrict__ W1pack, const _Float16* __restrict__ W2pack,
    float* __restrict__ acc_out) {

    __shared__ __align__(16) _Float16 sA[16][DM];        // pair_emb tile (f16, A layout)
    __shared__ __align__(16) _Float16 sH[3][16][256];    // relu hidden per head
    __shared__ float sOut[3][16][4];                     // head outputs per pair
    __shared__ float sD[16], sClose[16], sHyd[16], sHb[16], sScale[16];

    const int t = threadIdx.x;
    const int tile = blockIdx.x;                 // 0 .. NB*NN*16-1
    const int JT = NN / 16;
    const int b  = tile / (NN * JT);
    const int i  = (tile / JT) % NN;
    const int j0 = (tile % JT) * 16;
    const int w  = t >> 5, L = t & 31;
    const int r  = L & 15, hi = L >> 4;

    // prefetch this wave's weight stream (overlaps with LN prologue)
    for (int q = 0; q < 6; ++q)
        __builtin_prefetch(W1pack + ((size_t)(w * 6 + q) * 4) * 512, 0, 1);

    // ---- stage 0: per-pair scalars (16 threads) ----
    if (t < 16) {
        int p = t, j = j0 + p;
        int ti = node_type[b * NN + i]; if (ti >= NTYPES) ti -= NTYPES;
        int tj = node_type[b * NN + j]; if (tj >= NTYPES) tj -= NTYPES;
        int pt = ti * NTYPES + tj;
        float vdw = vdw_t[pt];
        float dd  = Dm[((size_t)(b * NN + i)) * NN + j] - vdw;
        float pm  = pair_mask[((size_t)(b * NN + i)) * NN + j] ? 1.f : 0.f;
        sD[p]     = dd;
        sClose[p] = ((dd < 4.0f) && (pm > 0.5f)) ? 1.f : 0.f;
        sHyd[p]   = (hydro_t[pt] > 0.5f) ? 1.f : 0.f;
        sHb[p]    = (hbond_t[pt] > 0.5f) ? 1.f : 0.f;
        sScale[p] = ((dd > 1.5f) ? 1.f : 0.f) * (1.f + ((dd < 2.0f) ? 1.f : 0.f));
    }
    __syncthreads();

    // ---- stage A: build pair_emb tile (16 pairs x 128 feats), 16 threads/pair ----
    {
        int p = t >> 4, s = t & 15;
        int j = j0 + p;
        const float* r1 = pair_feats + ((size_t)((b * 257 + 1 + i) * 257 + (1 + j))) * DM + s * 8;
        const float* r2 = pair_feats + ((size_t)((b * 257 + 1 + j) * 257 + (1 + i))) * DM + s * 8;
        f4 x1a = *(const f4*)r1,  x1b = *(const f4*)(r1 + 4);
        f4 x2a = *(const f4*)r2,  x2b = *(const f4*)(r2 + 4);
        float x1[8], x2[8];
        for (int k = 0; k < 4; ++k) { x1[k] = x1a[k]; x1[k+4] = x1b[k]; x2[k] = x2a[k]; x2[k+4] = x2b[k]; }

        float s1 = 0, q1 = 0, s2 = 0, q2 = 0;
        for (int k = 0; k < 8; ++k) { s1 += x1[k]; q1 += x1[k]*x1[k]; s2 += x2[k]; q2 += x2[k]*x2[k]; }
        s1 = redsum16(s1); q1 = redsum16(q1); s2 = redsum16(s2); q2 = redsum16(q2);
        float m1 = s1 * (1.f/DM), m2 = s2 * (1.f/DM);
        float rs1 = rsqrtf(q1 * (1.f/DM) - m1*m1 + 1e-6f);
        float rs2 = rsqrtf(q2 * (1.f/DM) - m2*m2 + 1e-6f);

        f4 pga = *(const f4*)(pg + s*8), pgb = *(const f4*)(pg + s*8 + 4);
        f4 pba = *(const f4*)(pb + s*8), pbb = *(const f4*)(pb + s*8 + 4);
        const float* nfi = nf + ((size_t)(b * NN + i)) * DM + s * 8;
        const float* nfj = nf + ((size_t)(b * NN + j)) * DM + s * 8;
        f4 nia = *(const f4*)nfi, nib = *(const f4*)(nfi + 4);
        f4 nja = *(const f4*)nfj, njb = *(const f4*)(nfj + 4);

        float e[8];
        for (int k = 0; k < 4; ++k) {
            float g0 = pga[k], b0 = pba[k], g1 = pgb[k], b1 = pbb[k];
            float pf0 = 0.5f * (((x1[k]   - m1)*rs1*g0 + b0) + ((x2[k]   - m2)*rs2*g0 + b0));
            float pf1 = 0.5f * (((x1[k+4] - m1)*rs1*g1 + b1) + ((x2[k+4] - m2)*rs2*g1 + b1));
            e[k]   = nia[k] * nja[k] + pf0;
            e[k+4] = nib[k] * njb[k] + pf1;
        }
        float se = 0, qe = 0;
        for (int k = 0; k < 8; ++k) { se += e[k]; qe += e[k]*e[k]; }
        se = redsum16(se); qe = redsum16(qe);
        float me = se * (1.f/DM);
        float rse = rsqrtf(qe * (1.f/DM) - me*me + 1e-6f);

        f4 fga = *(const f4*)(fg + s*8), fgb = *(const f4*)(fg + s*8 + 4);
        f4 fba = *(const f4*)(fb + s*8), fbb = *(const f4*)(fb + s*8 + 4);
        float scl = sScale[t >> 4];
        H8 st;
        for (int k = 0; k < 4; ++k) {
            st.h[k]   = (_Float16)(((e[k]   - me) * rse * fga[k] + fba[k]) * scl);
            st.h[k+4] = (_Float16)(((e[k+4] - me) * rse * fgb[k] + fbb[k]) * scl);
        }
        *(f4*)&sA[p][s * 8] = st.q;
    }
    __syncthreads();

    // ---- GEMM1: [16x128] @ [128 x 768] (3 heads fused), 8 waves x 6 n-tiles ----
    v8f acc[6];
    for (int q = 0; q < 6; ++q) for (int e = 0; e < 8; ++e) acc[q][e] = 0.f;

    for (int kb = 0; kb < 4; ++kb) {
        V16 af;
        const _Float16* ap = &sA[r][kb * 32 + hi * 16];
        af.q[0] = *(const f4*)ap;
        af.q[1] = *(const f4*)(ap + 8);
        for (int q = 0; q < 6; ++q) {
            int ntg = w * 6 + q;
            const _Float16* bp = W1pack + ((size_t)ntg * 4 + kb) * 512 + L * 16;
            V16 bf;
            bf.q[0] = *(const f4*)bp;
            bf.q[1] = *(const f4*)(bp + 8);
            acc[q] = __builtin_amdgcn_wmma_f32_16x16x32_f16(
                false, af.v, false, bf.v, (short)0, acc[q], false, false);
        }
    }

    // bias + ReLU -> sH (f16)
    for (int q = 0; q < 6; ++q) {
        int ntg = w * 6 + q;
        int hd  = ntg >> 4;
        int col = (ntg & 15) * 16 + r;
        const float* b1h = hd == 0 ? b1m : (hd == 1 ? b1s : b1p);
        float bias = b1h[col];
        for (int v = 0; v < 8; ++v) {
            float val = fmaxf(acc[q][v] + bias, 0.f);
            sH[hd][v + hi * 8][col] = (_Float16)val;
        }
    }
    __syncthreads();

    // ---- GEMM2: per head, [16x256] @ [256 x 16(pad 4)], waves 0..2 ----
    if (w < 3) {
        v8f a2;
        for (int e = 0; e < 8; ++e) a2[e] = 0.f;
        for (int kb = 0; kb < 8; ++kb) {
            V16 af;
            const _Float16* ap = &sH[w][r][kb * 32 + hi * 16];
            af.q[0] = *(const f4*)ap;
            af.q[1] = *(const f4*)(ap + 8);
            const _Float16* bp = W2pack + ((size_t)w * 8 + kb) * 512 + L * 16;
            V16 bf;
            bf.q[0] = *(const f4*)bp;
            bf.q[1] = *(const f4*)(bp + 8);
            a2 = __builtin_amdgcn_wmma_f32_16x16x32_f16(
                false, af.v, false, bf.v, (short)0, a2, false, false);
        }
        if (r < 4) {
            const float* b2h = w == 0 ? b2m : (w == 1 ? b2s : b2p);
            float b2v = b2h[r];
            for (int v = 0; v < 8; ++v) sOut[w][v + hi * 8][r] = a2[v] + b2v;
        }
    }
    __syncthreads();

    // ---- MDN epilogue: 16 threads, one pair each ----
    if (t < 16) {
        int p = t;
        float dd = sD[p], hyd = sHyd[p], hb = sHb[p], close = sClose[p];
        float loglik[4], pis[4];
        float pv[4];
        for (int k = 0; k < 4; ++k) {
            float mo = sOut[0][p][k], so = sOut[1][p][k];
            float mc = fmaxf(eluf(mo) + 1.f, 0.1f);
            float sg = fmaxf(eluf(so) + 1.f + 1e-6f, 1e-6f);
            float z  = (dd - mc) / sg;
            float lg = -0.5f * z * z - __logf(sg) - HALF_LOG_2PI;
            loglik[k] = (k == 2) ? lg * hyd : ((k == 3) ? lg * hb : lg);
            float msk = (k == 2) ? (hyd > 0.5f ? 0.f : NEG_INF)
                      : (k == 3) ? (hb  > 0.5f ? 0.f : NEG_INF) : 0.f;
            pv[k] = sOut[2][p][k] + msk;
        }
        float mx = fmaxf(fmaxf(pv[0], pv[1]), fmaxf(pv[2], pv[3]));
        float sme = 0.f;
        for (int k = 0; k < 4; ++k) { pis[k] = __expf(pv[k] - mx); sme += pis[k]; }
        float inv = 1.f / sme;
        for (int k = 0; k < 4; ++k) pis[k] = pis[k] * inv + 1e-9f;

        float lp[4], mx2 = -3.4e38f;
        for (int k = 0; k < 4; ++k) { lp[k] = __logf(pis[k] + 1e-9f) + loglik[k]; mx2 = fmaxf(mx2, lp[k]); }
        float es = 0.f, ps = 0.f;
        for (int k = 0; k < 4; ++k) { es += __expf(lp[k] - mx2); ps += pis[k] * __expf(loglik[k]); }
        float ll  = (mx2 + __logf(es)) * close;
        float neg = __logf(fmaxf(1.f - ps, 1e-9f)) * close;

        float ym = (yv[b] >= 0.f) ? 1.f : 0.f;
        float ce   = -(ym * ll + (1.f - ym) * neg);
        float pos  = -(ym * ll);
        float coll = fmaxf(2.0f - dd, 0.f);
        float mind = fmaxf(1.5f - dd, 0.f);

        ce = redsum16(ce); pos = redsum16(pos); coll = redsum16(coll); mind = redsum16(mind);
        if (t == 0) {
            atomicAdd(&acc_out[0], ce);
            atomicAdd(&acc_out[1], pos);
            atomicAdd(&acc_out[2], coll);
            atomicAdd(&acc_out[3], mind);
        }
    }
}

__global__ void finalize_kernel(const float* __restrict__ acc, float* __restrict__ out) {
    const float invN = 1.0f / (float)TOTAL_PAIRS;
    out[0] = acc[0] * invN + 0.1f * (acc[2] * invN + acc[3] * invN);
    out[1] = acc[1] * invN;
}

extern "C" void kernel_launch(void* const* d_in, const int* in_sizes, int n_in,
                              void* d_out, int out_size, void* d_ws, size_t ws_size,
                              hipStream_t stream) {
    const float* node_feats = (const float*)d_in[0];
    const float* pair_feats = (const float*)d_in[1];
    const float* Dm         = (const float*)d_in[2];
    const float* yv         = (const float*)d_in[3];
    const int*   node_type  = (const int*)d_in[4];
    const unsigned char* pair_mask = (const unsigned char*)d_in[5];
    const float* pg = (const float*)d_in[6],  *pb = (const float*)d_in[7];
    const float* ng = (const float*)d_in[8],  *nb = (const float*)d_in[9];
    const float* fg = (const float*)d_in[10], *fb = (const float*)d_in[11];
    const float* W1m = (const float*)d_in[12], *b1m = (const float*)d_in[13];
    const float* W2m = (const float*)d_in[14], *b2m = (const float*)d_in[15];
    const float* W1s = (const float*)d_in[16], *b1s = (const float*)d_in[17];
    const float* W2s = (const float*)d_in[18], *b2s = (const float*)d_in[19];
    const float* W1p = (const float*)d_in[20], *b1p = (const float*)d_in[21];
    const float* W2p = (const float*)d_in[22], *b2p = (const float*)d_in[23];
    const float* vdw_t   = (const float*)d_in[24];
    const float* hydro_t = (const float*)d_in[25];
    const float* hbond_t = (const float*)d_in[26];

    // workspace layout
    float*    acc    = (float*)d_ws;                                  // 8 floats (32 B)
    float*    nf     = (float*)((char*)d_ws + 256);                   // NB*NN*DM f32 = 512 KB
    _Float16* W1pack = (_Float16*)((char*)d_ws + 256 + (size_t)NB*NN*DM*4);
    _Float16* W2pack = W1pack + (size_t)3 * 16 * 4 * 512;             // 96 K halves

    hipMemsetAsync(acc, 0, 32, stream);
    nf_kernel<<<NB * NN / 8, 256, 0, stream>>>(node_feats, ng, nb, nf);
    pack_kernel<<<27, 256, 0, stream>>>(W1m, W1s, W1p, W2m, W2s, W2p, W1pack, W2pack);
    main_kernel<<<NB * NN * (NN / 16), 256, 0, stream>>>(
        pair_feats, Dm, yv, node_type, pair_mask,
        pg, pb, fg, fb,
        b1m, b1s, b1p, b2m, b2s, b2p,
        vdw_t, hydro_t, hbond_t,
        nf, W1pack, W2pack, acc);
    finalize_kernel<<<1, 1, 0, stream>>>(acc, (float*)d_out);
}
